// DeepseekMoE_89773406421361
// MI455X (gfx1250) — matrix operand order repaired and checked by reading, not compiled, not run on hardware
//
#include <hip/hip_runtime.h>
#include <hip/hip_bf16.h>

#define T_TOK 1024
#define HDIM  2048
#define IDIM  1408
#define NEXP  16
#define NE_ALL 18   // 16 routed + 2 shared halves (width IDIM each, weight 1.0)

typedef __attribute__((ext_vector_type(16))) __bf16 v16bf;
typedef __attribute__((ext_vector_type(8)))  float  v8f;

union Frag16 { v16bf v; uint4 q[2]; };

// ---- CDNA5 async global->LDS path (guarded; falls back to plain LDS copy) ---
#if defined(__has_builtin)
#if __has_builtin(__builtin_amdgcn_global_load_async_to_lds_b128) && \
    __has_builtin(__builtin_amdgcn_s_wait_asynccnt)
#define USE_ASYNC_LDS 1
#endif
#endif

#ifdef USE_ASYNC_LDS
typedef int v4i_vs __attribute__((vector_size(16)));
typedef __attribute__((address_space(1))) v4i_vs* gbl_v4i;
typedef __attribute__((address_space(3))) v4i_vs* lds_v4i;
typedef __attribute__((address_space(1))) void*   gbl_vp;
typedef __attribute__((address_space(3))) void*   lds_vp;
#endif

__device__ __forceinline__ void copy16_g2l(const __bf16* g, __bf16* l) {
#ifdef USE_ASYNC_LDS
    __builtin_amdgcn_global_load_async_to_lds_b128(
        (gbl_v4i)(gbl_vp)g, (lds_v4i)(lds_vp)l, 0, 0);
#else
    *(uint4*)l = *(const uint4*)g;
#endif
}
__device__ __forceinline__ void wait_async_g2l() {
#ifdef USE_ASYNC_LDS
    __builtin_amdgcn_s_wait_asynccnt(0);
#endif
}

__device__ __forceinline__ float fast_silu(float g) {
#if defined(__HIP_DEVICE_COMPILE__)
    return g * __builtin_amdgcn_rcpf(1.0f + __expf(-g));
#else
    return g;
#endif
}

// ---------------- Kernel 0: fp32 -> bf16 copy of activations -----------------
__global__ void k_cvt_bf16(const float* __restrict__ x, __bf16* __restrict__ y, int n) {
    int i = (blockIdx.x * blockDim.x + threadIdx.x) * 4;
    if (i < n) {
        float4 v = *(const float4*)(x + i);
        y[i + 0] = (__bf16)v.x; y[i + 1] = (__bf16)v.y;
        y[i + 2] = (__bf16)v.z; y[i + 3] = (__bf16)v.w;
    }
}

// ---------------- Kernel 1: gate logits + softmax + top-4 -> combine[T,18] ---
__global__ void k_gate(const float* __restrict__ x, const float* __restrict__ gw,
                       float* __restrict__ combine) {
    __shared__ float lg[16][17];
    const int tl  = threadIdx.x >> 4;   // local token 0..15
    const int e   = threadIdx.x & 15;   // expert 0..15
    const int tok = blockIdx.x * 16 + tl;
    const float4* xr = (const float4*)(x + (size_t)tok * HDIM);
    const float4* wr = (const float4*)(gw + (size_t)e * HDIM);
    float acc = 0.f;
    for (int h = 0; h < HDIM / 4; ++h) {
        float4 a = xr[h], b = wr[h];
        acc += a.x * b.x + a.y * b.y + a.z * b.z + a.w * b.w;
    }
    lg[tl][e] = acc;
    __syncthreads();
    if (threadIdx.x < 16) {
        const int t    = threadIdx.x;
        const int tok2 = blockIdx.x * 16 + t;
        float p[16], m = -1e30f;
        for (int j = 0; j < 16; ++j) m = fmaxf(m, lg[t][j]);
        float s = 0.f;
        for (int j = 0; j < 16; ++j) { p[j] = __expf(lg[t][j] - m); s += p[j]; }
        const float inv = 1.f / s;
        float row[NE_ALL];
        for (int j = 0; j < NE_ALL; ++j) row[j] = 0.f;
        float tmp[16];
        for (int j = 0; j < 16; ++j) tmp[j] = p[j];
        for (int k = 0; k < 4; ++k) {          // top-4, first-occurrence ties
            int bi = 0; float bv = tmp[0];
            for (int j = 1; j < 16; ++j) if (tmp[j] > bv) { bv = tmp[j]; bi = j; }
            row[bi] = bv * inv;                // norm_topk_prob=False
            tmp[bi] = -1.f;
        }
        row[16] = 1.f; row[17] = 1.f;          // shared expert halves always on
        for (int j = 0; j < NE_ALL; ++j) combine[(size_t)tok2 * NE_ALL + j] = row[j];
    }
}

// ---------------- Kernel 2: per-expert gate/up GEMM + SwiGLU + weight --------
// Block: 256 thr (8 waves), tile 64(M tokens) x 128(N intermediate), K-stage 64.
__global__ __launch_bounds__(256)
void k_gemm1(const __bf16* __restrict__ Xb,
             const float* __restrict__ w_gate, const float* __restrict__ w_up,
             const float* __restrict__ sw_gate, const float* __restrict__ sw_up,
             const float* __restrict__ combine, __bf16* __restrict__ Hb) {
    __shared__ __bf16 Ash[64 * 72];     // 64x64 tile, stride 72 -> conflict-free
    __shared__ __bf16 Bgs[128 * 72];    // stored n-major (transposed)
    __shared__ __bf16 Bus[128 * 72];

    const int e  = blockIdx.z;
    const int nb = blockIdx.x * 128;
    const int m0 = blockIdx.y * 64;

    const float *Wg, *Wu; int ld;
    if (e < NEXP) { size_t o = (size_t)e * HDIM * IDIM; Wg = w_gate + o; Wu = w_up + o; ld = IDIM; }
    else          { int o = (e - NEXP) * IDIM; Wg = sw_gate + o; Wu = sw_up + o; ld = 2 * IDIM; }

    const int tid  = threadIdx.x;
    const int wave = tid >> 5, lane = tid & 31;
    const int wm = wave & 1, wn = wave >> 1;   // 2 x 4 wave grid
    const int l15 = lane & 15, kh = lane >> 4;
    // A loader: two 16B segs/thread over 64 rows x 128B
    const int ar0 = tid >> 3,         ac0 = (tid & 7) * 8;
    const int ar1 = (tid + 256) >> 3, ac1 = ((tid + 256) & 7) * 8;
    // B loader: col + 32-deep k strip
    const int bn = tid & 127, bk = (tid >> 7) * 32;

    v8f accG[2][2] = {};
    v8f accU[2][2] = {};

    for (int k0 = 0; k0 < HDIM; k0 += 64) {
        // A tile: async global->LDS (bf16, no conversion needed)
        copy16_g2l(&Xb[(size_t)(m0 + ar0) * HDIM + k0 + ac0], &Ash[ar0 * 72 + ac0]);
        copy16_g2l(&Xb[(size_t)(m0 + ar1) * HDIM + k0 + ac1], &Ash[ar1 * 72 + ac1]);
        // B tiles: fp32 strided-coalesced load, cvt->bf16, store n-major
        __bf16 tg[32] __attribute__((aligned(16)));
        __bf16 tu[32] __attribute__((aligned(16)));
        const float* pg = Wg + (size_t)(k0 + bk) * ld + nb + bn;
        const float* pu = Wu + (size_t)(k0 + bk) * ld + nb + bn;
        __builtin_prefetch(pg + (size_t)64 * ld, 0, 1);   // next K stage
        __builtin_prefetch(pu + (size_t)64 * ld, 0, 1);
#pragma unroll
        for (int kk = 0; kk < 32; ++kk) {
            tg[kk] = (__bf16)pg[(size_t)kk * ld];
            tu[kk] = (__bf16)pu[(size_t)kk * ld];
        }
#pragma unroll
        for (int i = 0; i < 4; ++i) {
            *(uint4*)&Bgs[bn * 72 + bk + 8 * i] = *(uint4*)&tg[8 * i];
            *(uint4*)&Bus[bn * 72 + bk + 8 * i] = *(uint4*)&tu[8 * i];
        }
        wait_async_g2l();
        __syncthreads();

#pragma unroll
        for (int kt = 0; kt < 2; ++kt) {       // two K=32 WMMA steps per stage
            Frag16 a[2], bg[2], bu[2];
#pragma unroll
            for (int sm = 0; sm < 2; ++sm) {   // A frag: lane m=l15, K split by kh
                const __bf16* p = &Ash[(wm * 32 + sm * 16 + l15) * 72 + kt * 32];
                a[sm].q[0] = *(const uint4*)&p[kh * 8];
                a[sm].q[1] = *(const uint4*)&p[16 + kh * 8];
            }
#pragma unroll
            for (int sn = 0; sn < 2; ++sn) {   // B frag: lane n=l15, K split by kh
                const __bf16* p = &Bgs[(wn * 32 + sn * 16 + l15) * 72 + kt * 32];
                bg[sn].q[0] = *(const uint4*)&p[kh * 8];
                bg[sn].q[1] = *(const uint4*)&p[16 + kh * 8];
                const __bf16* q = &Bus[(wn * 32 + sn * 16 + l15) * 72 + kt * 32];
                bu[sn].q[0] = *(const uint4*)&q[kh * 8];
                bu[sn].q[1] = *(const uint4*)&q[16 + kh * 8];
            }
#pragma unroll
            for (int sm = 0; sm < 2; ++sm)
#pragma unroll
                for (int sn = 0; sn < 2; ++sn) {
                    accG[sm][sn] = __builtin_amdgcn_wmma_f32_16x16x32_bf16(
                        false, a[sm].v, false, bg[sn].v, (short)0, accG[sm][sn], false, false);
                    accU[sm][sn] = __builtin_amdgcn_wmma_f32_16x16x32_bf16(
                        false, a[sm].v, false, bu[sn].v, (short)0, accU[sm][sn], false, false);
                }
        }
        __syncthreads();
    }

    // epilogue: h = silu(g) * u * combine  (fold routing weight into intermediate)
    __bf16* He = Hb + (size_t)e * T_TOK * IDIM;
#pragma unroll
    for (int sm = 0; sm < 2; ++sm) {
        const int tokb = m0 + wm * 32 + sm * 16 + kh * 8;   // C layout: m = kh*8 + j
        float cw[8];
#pragma unroll
        for (int j = 0; j < 8; ++j) cw[j] = combine[(size_t)(tokb + j) * NE_ALL + e];
#pragma unroll
        for (int sn = 0; sn < 2; ++sn) {
            const int col = nb + wn * 32 + sn * 16 + l15;
#pragma unroll
            for (int j = 0; j < 8; ++j) {
                float g = accG[sm][sn][j], u = accU[sm][sn][j];
                He[(size_t)(tokb + j) * IDIM + col] = (__bf16)(fast_silu(g) * u * cw[j]);
            }
        }
    }
}

// ---------------- Kernel 3: down-proj accumulate over all 18 experts ---------
__global__ __launch_bounds__(256)
void k_gemm2(const __bf16* __restrict__ Hb,
             const float* __restrict__ w_down, const float* __restrict__ sw_down,
             float* __restrict__ out) {
    __shared__ __bf16 Ash[64 * 72];
    __shared__ __bf16 Bds[128 * 72];
    const int nb = blockIdx.x * 128;
    const int m0 = blockIdx.y * 64;
    const int tid  = threadIdx.x;
    const int wave = tid >> 5, lane = tid & 31;
    const int wm = wave & 1, wn = wave >> 1;
    const int l15 = lane & 15, kh = lane >> 4;
    const int ar0 = tid >> 3,         ac0 = (tid & 7) * 8;
    const int ar1 = (tid + 256) >> 3, ac1 = ((tid + 256) & 7) * 8;
    const int bn = tid & 127, bk = (tid >> 7) * 32;

    v8f acc[2][2] = {};

    for (int e = 0; e < NE_ALL; ++e) {
        const __bf16* Ae = Hb + (size_t)e * T_TOK * IDIM;
        const float*  Wd = (e < NEXP) ? (w_down + (size_t)e * IDIM * HDIM)
                                      : (sw_down + (size_t)(e - NEXP) * IDIM * HDIM);
        for (int k0 = 0; k0 < IDIM; k0 += 64) {
            copy16_g2l(&Ae[(size_t)(m0 + ar0) * IDIM + k0 + ac0], &Ash[ar0 * 72 + ac0]);
            copy16_g2l(&Ae[(size_t)(m0 + ar1) * IDIM + k0 + ac1], &Ash[ar1 * 72 + ac1]);
            __bf16 td[32] __attribute__((aligned(16)));
            const float* pd = Wd + (size_t)(k0 + bk) * HDIM + nb + bn;
            __builtin_prefetch(pd + (size_t)64 * HDIM, 0, 1);
#pragma unroll
            for (int kk = 0; kk < 32; ++kk) td[kk] = (__bf16)pd[(size_t)kk * HDIM];
#pragma unroll
            for (int i = 0; i < 4; ++i)
                *(uint4*)&Bds[bn * 72 + bk + 8 * i] = *(uint4*)&td[8 * i];
            wait_async_g2l();
            __syncthreads();

#pragma unroll
            for (int kt = 0; kt < 2; ++kt) {
                Frag16 a[2], b[2];
#pragma unroll
                for (int sm = 0; sm < 2; ++sm) {
                    const __bf16* p = &Ash[(wm * 32 + sm * 16 + l15) * 72 + kt * 32];
                    a[sm].q[0] = *(const uint4*)&p[kh * 8];
                    a[sm].q[1] = *(const uint4*)&p[16 + kh * 8];
                }
#pragma unroll
                for (int sn = 0; sn < 2; ++sn) {
                    const __bf16* p = &Bds[(wn * 32 + sn * 16 + l15) * 72 + kt * 32];
                    b[sn].q[0] = *(const uint4*)&p[kh * 8];
                    b[sn].q[1] = *(const uint4*)&p[16 + kh * 8];
                }
#pragma unroll
                for (int sm = 0; sm < 2; ++sm)
#pragma unroll
                    for (int sn = 0; sn < 2; ++sn)
                        acc[sm][sn] = __builtin_amdgcn_wmma_f32_16x16x32_bf16(
                            false, a[sm].v, false, b[sn].v, (short)0, acc[sm][sn], false, false);
            }
            __syncthreads();
        }
    }

#pragma unroll
    for (int sm = 0; sm < 2; ++sm) {
        const int tokb = m0 + wm * 32 + sm * 16 + kh * 8;
#pragma unroll
        for (int sn = 0; sn < 2; ++sn) {
            const int col = nb + wn * 32 + sn * 16 + l15;
#pragma unroll
            for (int j = 0; j < 8; ++j)
                out[(size_t)(tokb + j) * HDIM + col] = acc[sm][sn][j];
        }
    }
}

extern "C" void kernel_launch(void* const* d_in, const int* in_sizes, int n_in,
                              void* d_out, int out_size, void* d_ws, size_t ws_size,
                              hipStream_t stream) {
    const float* x       = (const float*)d_in[0];
    const float* gate_w  = (const float*)d_in[1];
    const float* w_gate  = (const float*)d_in[2];
    const float* w_up    = (const float*)d_in[3];
    const float* w_down  = (const float*)d_in[4];
    const float* sw_gate = (const float*)d_in[5];
    const float* sw_up   = (const float*)d_in[6];
    const float* sw_down = (const float*)d_in[7];
    float* out = (float*)d_out;

    // workspace: Xb (bf16 X) | combine [T,18] f32 | Hb (bf16 weighted intermediates)
    char* ws = (char*)d_ws;
    __bf16* Xb = (__bf16*)ws;
    size_t off = (size_t)T_TOK * HDIM * 2;                  // 4 MB
    float* combine = (float*)(ws + off);
    off += (size_t)T_TOK * NE_ALL * 4;                      // 73728 B (16B aligned)
    __bf16* Hb = (__bf16*)(ws + off);                       // 18*1024*1408*2 ~= 49.5 MB

    k_cvt_bf16<<<(T_TOK * HDIM) / (256 * 4), 256, 0, stream>>>(x, Xb, T_TOK * HDIM);
    k_gate<<<T_TOK / 16, 256, 0, stream>>>(x, gate_w, combine);
    dim3 g1(IDIM / 128, T_TOK / 64, NE_ALL);
    k_gemm1<<<g1, 256, 0, stream>>>(Xb, w_gate, w_up, sw_gate, sw_up, combine, Hb);
    dim3 g2(HDIM / 128, T_TOK / 64);
    k_gemm2<<<g2, 256, 0, stream>>>(Hb, w_down, sw_down, out);
}
